// RNet_31817117729551
// MI455X (gfx1250) — matrix operand order repaired and checked by reading, not compiled, not run on hardware
//
#include <hip/hip_runtime.h>
#include <math.h>

// ---------------------------------------------------------------------------
// CDNA5 WMMA helpers (wave32, v_wmma_f32_16x16x32_bf16)
// ---------------------------------------------------------------------------
typedef __attribute__((ext_vector_type(16))) __bf16 v16bf;
typedef __attribute__((ext_vector_type(8)))  float  v8f;

#define WMMA_BF16(a, b, c) \
  __builtin_amdgcn_wmma_f32_16x16x32_bf16(false, (a), false, (b), (short)0, (c), false, false)

__device__ __forceinline__ v8f vzero8() {
  v8f z = {0.f, 0.f, 0.f, 0.f, 0.f, 0.f, 0.f, 0.f};
  return z;
}
__device__ __forceinline__ float sigmf(float x) { return 1.f / (1.f + __expf(-x)); }

// A-fragment (16x32 bf16) from LDS, row-major [mrow][k], leading dim ld.
// lanes 0-15: rows M=0..15, slots 0..15=K+0..15;
// lanes 16-31: same rows, slots 0..15=K+16..31.
__device__ __forceinline__ v16bf ldsA(const __bf16* base, int ld, int mrow, int kc, int lane) {
  v16bf a;
  const __bf16* p = base + mrow * ld + kc + ((lane < 16) ? 0 : 16);
#pragma unroll
  for (int i = 0; i < 16; ++i) a[i] = p[i];
  return a;
}

// B-fragment (32x16) from LDS bf16 matrix with leading dim ld.
// lanes 0-15: slot j = [kc+j][ncol]; lanes 16-31: slot j = [kc+16+j][ncol].
__device__ __forceinline__ v16bf ldB_lds(const __bf16* Bs, int ld, int kc, int ncol, int lane) {
  v16bf b;
  const __bf16* p = Bs + (size_t)(kc + ((lane < 16) ? 0 : 16)) * ld + ncol;
#pragma unroll
  for (int j = 0; j < 16; ++j) b[j] = p[(size_t)j * ld];
  return b;
}

// B-fragment from a global f32 matrix W[krows x ncols] (ld=ldw), zero-padded.
__device__ __forceinline__ v16bf ldB_global(const float* W, int ldw, int krows, int ncols,
                                            int kc, int ncol, int lane) {
  v16bf b;
  int kb = kc + ((lane < 16) ? 0 : 16);
#pragma unroll
  for (int j = 0; j < 16; ++j) {
    int kk = kb + j;
    float w = (kk < krows && ncol < ncols) ? W[(size_t)kk * ldw + ncol] : 0.f;
    b[j] = (__bf16)w;
  }
  return b;
}

// ---------------------------------------------------------------------------
// Generic GEMM: C[R x N] = A[R x K] @ W[K x N] (+ bias).  f32 in, bf16 WMMA, f32 out.
// Block = 256 threads (8 waves), 64x64 tile, each wave computes two 16x16 tiles.
// Fast-path staging (no per-element masks) when the chunk is fully in-bounds.
// ---------------------------------------------------------------------------
__global__ __launch_bounds__(256) void gemm_bias_kernel(
    const float* __restrict__ A, const float* __restrict__ W,
    const float* __restrict__ bias, float* __restrict__ C,
    int R, int K, int N) {
  __shared__ __bf16 As[64 * 40];
  __shared__ __bf16 Bs[32 * 72];
  int tn = (N + 63) >> 6;
  int bm = blockIdx.x / tn, bn = blockIdx.x % tn;
  int row0 = bm * 64, col0 = bn * 64;
  int tid = threadIdx.x, wave = tid >> 5, lane = tid & 31;
  v8f acc0 = vzero8(), acc1 = vzero8();
  for (int kc = 0; kc < K; kc += 32) {
    bool fullK = (kc + 32 <= K);
    if (fullK && (row0 + 64 <= R)) {
      for (int idx = tid; idx < 64 * 8; idx += 256) {
        int r = idx >> 3, k4 = (idx & 7) << 2;
        const float* p = A + (size_t)(row0 + r) * K + kc + k4;
        As[r * 40 + k4 + 0] = (__bf16)p[0];
        As[r * 40 + k4 + 1] = (__bf16)p[1];
        As[r * 40 + k4 + 2] = (__bf16)p[2];
        As[r * 40 + k4 + 3] = (__bf16)p[3];
      }
    } else {
      for (int idx = tid; idx < 64 * 32; idx += 256) {
        int r = idx >> 5, k = idx & 31;
        float val = ((row0 + r) < R && (kc + k) < K) ? A[(size_t)(row0 + r) * K + kc + k] : 0.f;
        As[r * 40 + k] = (__bf16)val;
      }
    }
    if (fullK && (col0 + 64 <= N)) {
      for (int idx = tid; idx < 32 * 16; idx += 256) {
        int k = idx >> 4, c4 = (idx & 15) << 2;
        const float* p = W + (size_t)(kc + k) * N + col0 + c4;
        Bs[k * 72 + c4 + 0] = (__bf16)p[0];
        Bs[k * 72 + c4 + 1] = (__bf16)p[1];
        Bs[k * 72 + c4 + 2] = (__bf16)p[2];
        Bs[k * 72 + c4 + 3] = (__bf16)p[3];
      }
    } else {
      for (int idx = tid; idx < 32 * 64; idx += 256) {
        int k = idx >> 6, cc = idx & 63;
        float val = ((kc + k) < K && (col0 + cc) < N) ? W[(size_t)(kc + k) * N + col0 + cc] : 0.f;
        Bs[k * 72 + cc] = (__bf16)val;
      }
    }
    __syncthreads();
#pragma unroll
    for (int tt = 0; tt < 2; ++tt) {
      int tile = wave + tt * 8;
      int tm = (tile >> 2) * 16, tnn = (tile & 3) * 16;
      v16bf a = ldsA(As, 40, tm + (lane & 15), 0, lane);
      v16bf b = ldB_lds(Bs, 72, 0, tnn + (lane & 15), lane);
      if (tt == 0) acc0 = WMMA_BF16(a, b, acc0);
      else         acc1 = WMMA_BF16(a, b, acc1);
    }
    __syncthreads();
  }
#pragma unroll
  for (int tt = 0; tt < 2; ++tt) {
    int tile = wave + tt * 8;
    int tm = (tile >> 2) * 16, tnn = (tile & 3) * 16;
    v8f acc = tt ? acc1 : acc0;
    int col = col0 + tnn + (lane & 15);
    if (col < N) {
      float bv = bias ? bias[col] : 0.f;
#pragma unroll
      for (int r = 0; r < 8; ++r) {
        int row = row0 + tm + r + ((lane < 16) ? 0 : 8);
        if (row < R) C[(size_t)row * N + col] = acc[r] + bv;
      }
    }
  }
}

// ---------------------------------------------------------------------------
// Persistent GRU scan (pair): 2 blocks, each runs one direction.
// m (B,T,225) precomputed (= x@W + b).  U (75,225) = [Uz|Ur|Uh].
// GRU: z=sig(mz+h@Uz), r=sig(mr+h@Ur), hh=tanh(mh+(r*h)@Uh), h=z*h+(1-z)*hh.
// U is staged once into LDS (bf16, padded 96x240) so per-step WMMA B-fragments
// are ds_loads.  Dynamic LDS = 82,944 B (within gfx1250's 320KB/WGP).
// ---------------------------------------------------------------------------
__global__ __launch_bounds__(256) void gru_scan_pair_kernel(
    const float* m0, const float* U0, int rev0, int oa0, int c0,
    const float* m1, const float* U1, int rev1, int oa1, int c1,
    float* __restrict__ out, int T, int ostride) {
  const float* m_ = (blockIdx.x == 0) ? m0 : m1;
  const float* U  = (blockIdx.x == 0) ? U0 : U1;
  int rev  = (blockIdx.x == 0) ? rev0 : rev1;
  int oa   = (blockIdx.x == 0) ? oa0  : oa1;
  int col0 = (blockIdx.x == 0) ? c0   : c1;
  extern __shared__ char smem[];
  float*  h  = (float*)(smem);            // 32*80  f32   (10240 B)
  __bf16* hb = (__bf16*)(smem + 10240);   // 32*96  bf16  ( 6144 B)
  float*  g  = (float*)(smem + 16384);    // 32*160 f32   (20480 B)
  __bf16* Ub = (__bf16*)(smem + 36864);   // 96*240 bf16  (46080 B) -> 82944 total
  int tid = threadIdx.x, wave = tid >> 5, lane = tid & 31;
  for (int idx = tid; idx < 96 * 240; idx += 256) {
    int k = idx / 240, cc = idx % 240;
    float w = (k < 75 && cc < 225) ? U[(size_t)k * 225 + cc] : 0.f;
    Ub[idx] = (__bf16)w;
  }
  for (int i = tid; i < 32 * 96; i += 256) hb[i] = (__bf16)0.f;
  for (int i = tid; i < 32 * 80; i += 256) h[i] = 0.f;
  __syncthreads();
  for (int i = 0; i < T; ++i) {
    int t  = rev ? (T - 1 - i) : i;
    int to = oa ? t : i;
    // h -> bf16; prefetch next step's m rows (global_prefetch_b8)
    for (int idx = tid; idx < 32 * 75; idx += 256) {
      int b = idx / 75, j = idx % 75;
      hb[b * 96 + j] = (__bf16)h[b * 80 + j];
    }
    if (i + 1 < T && tid < 32) {
      int tn_ = rev ? (T - 2 - i) : (i + 1);
      __builtin_prefetch(m_ + ((size_t)(tid * T + tn_)) * 225, 0, 1);
    }
    __syncthreads();
    // g[:,0:160] = h @ U[:,0:160]  (Uz|Ur in cols 0..149)
    for (int tile = wave; tile < 20; tile += 8) {
      int mt = tile / 10, nt = tile % 10, ncol = nt * 16 + (lane & 15);
      v8f acc = vzero8();
      for (int kc = 0; kc < 96; kc += 32) {
        v16bf a = ldsA(hb, 96, mt * 16 + (lane & 15), kc, lane);
        v16bf b = ldB_lds(Ub, 240, kc, ncol, lane);
        acc = WMMA_BF16(a, b, acc);
      }
#pragma unroll
      for (int r = 0; r < 8; ++r) {
        int row = mt * 16 + r + ((lane < 16) ? 0 : 8);
        g[row * 160 + ncol] = acc[r];
      }
    }
    __syncthreads();
    // r, rh -> hb
    for (int idx = tid; idx < 32 * 75; idx += 256) {
      int b = idx / 75, j = idx % 75;
      const float* mt_ = m_ + ((size_t)(b * T + t)) * 225;
      float r = sigmf(mt_[75 + j] + g[b * 160 + 75 + j]);
      hb[b * 96 + j] = (__bf16)(r * h[b * 80 + j]);
    }
    __syncthreads();
    // g[:,80:160] = (r*h) @ Uh
    for (int tile = wave; tile < 10; tile += 8) {
      int mt = tile / 5, nt = tile % 5, ncol = nt * 16 + (lane & 15);
      v8f acc = vzero8();
      for (int kc = 0; kc < 96; kc += 32) {
        v16bf a = ldsA(hb, 96, mt * 16 + (lane & 15), kc, lane);
        v16bf b = ldB_lds(Ub, 240, kc, 150 + ncol, lane);
        acc = WMMA_BF16(a, b, acc);
      }
#pragma unroll
      for (int r = 0; r < 8; ++r) {
        int row = mt * 16 + r + ((lane < 16) ? 0 : 8);
        g[row * 160 + 80 + ncol] = acc[r];
      }
    }
    __syncthreads();
    // combine, write state + output
    for (int idx = tid; idx < 32 * 75; idx += 256) {
      int b = idx / 75, j = idx % 75;
      const float* mt_ = m_ + ((size_t)(b * T + t)) * 225;
      float z  = sigmf(mt_[j] + g[b * 160 + j]);
      float hh = tanhf(mt_[150 + j] + g[b * 160 + 80 + j]);
      float hn = z * h[b * 80 + j] + (1.f - z) * hh;
      h[b * 80 + j] = hn;
      out[((size_t)(b * T + to)) * ostride + col0 + j] = hn;
    }
    __syncthreads();
  }
}

// ---------------------------------------------------------------------------
// Question-attention GRU: fully sequential (attention depends on h).
// One persistent workgroup.  W_g1 (320x304 bf16) and WP_v (96x80 bf16) staged
// in LDS; dynamic LDS total 294,400 B (gfx1250 WGP supports 320KB).
// ---------------------------------------------------------------------------
__global__ __launch_bounds__(256) void qattn_scan_kernel(
    const float* __restrict__ uP, const float* __restrict__ uQ,
    const float* __restrict__ WPu, const float* __restrict__ WQuQ,
    const float* __restrict__ WPv, const float* __restrict__ vvec,
    const float* __restrict__ Wg1,
    const float* __restrict__ Wqa, const float* __restrict__ Uqa, const float* __restrict__ bqa,
    float* __restrict__ vP) {
  extern __shared__ char smem[];
  float*  h     = (float*)(smem);                 // 32*80  f32   (10240 B)
  __bf16* hb    = (__bf16*)(smem + 10240);        // 32*96  bf16  ( 6144 B)
  float*  g     = (float*)(smem + 16384);         // 32*160 f32   (20480 B)
  float*  sc    = (float*)(smem + 36864);         // 32*52  f32   ( 6656 B)
  __bf16* inpb  = (__bf16*)(smem + 43520);        // 32*320 bf16  (20480 B)
  __bf16* inp2b = (__bf16*)(smem + 64000);        // 32*320 bf16  (20480 B)
  __bf16* WPvb  = (__bf16*)(smem + 84480);        // 96*80  bf16  (15360 B)
  __bf16* Wg1b  = (__bf16*)(smem + 99840);        // 320*304 bf16 (194560 B) -> 294400
  int tid = threadIdx.x, wave = tid >> 5, lane = tid & 31;
  for (int idx = tid; idx < 96 * 80; idx += 256) {
    int k = idx / 80, cc = idx % 80;
    WPvb[idx] = (__bf16)((k < 75 && cc < 75) ? WPv[(size_t)k * 75 + cc] : 0.f);
  }
  for (int idx = tid; idx < 320 * 304; idx += 256) {
    int k = idx / 304, cc = idx % 304;
    Wg1b[idx] = (__bf16)((k < 300 && cc < 300) ? Wg1[(size_t)k * 300 + cc] : 0.f);
  }
  for (int i = tid; i < 32 * 96; i += 256) hb[i] = (__bf16)0.f;
  for (int i = tid; i < 32 * 80; i += 256) h[i] = 0.f;
  for (int i = tid; i < 32 * 320; i += 256) { inpb[i] = (__bf16)0.f; inp2b[i] = (__bf16)0.f; }
  __syncthreads();
  for (int t = 0; t < 400; ++t) {
    for (int idx = tid; idx < 32 * 75; idx += 256) {
      int b = idx / 75, j = idx % 75;
      hb[b * 96 + j] = (__bf16)h[b * 80 + j];
    }
    if (t + 1 < 400 && tid < 32) {
      __builtin_prefetch(uP + ((size_t)(tid * 400 + t + 1)) * 150, 0, 1);
      __builtin_prefetch(WPu + ((size_t)(tid * 400 + t + 1)) * 75, 0, 1);
    }
    __syncthreads();
    // s = h @ WP_v  -> g[:,0:80]
    for (int tile = wave; tile < 10; tile += 8) {
      int mt = tile / 5, nt = tile % 5, ncol = nt * 16 + (lane & 15);
      v8f acc = vzero8();
      for (int kc = 0; kc < 96; kc += 32) {
        v16bf a = ldsA(hb, 96, mt * 16 + (lane & 15), kc, lane);
        v16bf b = ldB_lds(WPvb, 80, kc, ncol, lane);
        acc = WMMA_BF16(a, b, acc);
      }
#pragma unroll
      for (int r = 0; r < 8; ++r) {
        int row = mt * 16 + r + ((lane < 16) ? 0 : 8);
        g[row * 160 + ncol] = acc[r];
      }
    }
    __syncthreads();
    for (int idx = tid; idx < 32 * 75; idx += 256) {
      int b = idx / 75, j = idx % 75;
      g[b * 160 + j] += WPu[((size_t)(b * 400 + t)) * 75 + j];
    }
    __syncthreads();
    // attention scores over question positions
    for (int idx = tid; idx < 32 * 50; idx += 256) {
      int b = idx / 50, mq = idx % 50;
      const float* wq = WQuQ + ((size_t)(b * 50 + mq)) * 75;
      float s = 0.f;
      for (int j = 0; j < 75; ++j) s += vvec[j] * tanhf(wq[j] + g[b * 160 + j]);
      sc[b * 52 + mq] = s;
    }
    __syncthreads();
    if (tid < 32) {
      float mx = -3.4e38f;
      for (int q = 0; q < 50; ++q) mx = fmaxf(mx, sc[tid * 52 + q]);
      float su = 0.f;
      for (int q = 0; q < 50; ++q) { float e = __expf(sc[tid * 52 + q] - mx); sc[tid * 52 + q] = e; su += e; }
      float inv = 1.f / su;
      for (int q = 0; q < 50; ++q) sc[tid * 52 + q] *= inv;
    }
    __syncthreads();
    // inp = [uP_t, c]
    for (int idx = tid; idx < 32 * 150; idx += 256) {
      int b = idx / 150, d = idx % 150;
      inpb[b * 320 + d] = (__bf16)uP[((size_t)(b * 400 + t)) * 150 + d];
    }
    for (int idx = tid; idx < 32 * 150; idx += 256) {
      int b = idx / 150, d = idx % 150;
      float c = 0.f;
      for (int q = 0; q < 50; ++q) c += sc[b * 52 + q] * uQ[((size_t)(b * 50 + q)) * 150 + d];
      inpb[b * 320 + 150 + d] = (__bf16)c;
    }
    __syncthreads();
    // gate: inp2 = sigmoid(inp @ W_g1) * inp   (W_g1 from LDS)
    for (int tile = wave; tile < 38; tile += 8) {
      int mt = tile / 19, nt = tile % 19, ncol = nt * 16 + (lane & 15);
      v8f acc = vzero8();
      for (int kc = 0; kc < 320; kc += 32) {
        v16bf a = ldsA(inpb, 320, mt * 16 + (lane & 15), kc, lane);
        v16bf b = ldB_lds(Wg1b, 304, kc, ncol, lane);
        acc = WMMA_BF16(a, b, acc);
      }
      if (ncol < 300) {
#pragma unroll
        for (int r = 0; r < 8; ++r) {
          int row = mt * 16 + r + ((lane < 16) ? 0 : 8);
          float s = sigmf(acc[r]);
          inp2b[row * 320 + ncol] = (__bf16)(s * (float)inpb[row * 320 + ncol]);
        }
      }
    }
    __syncthreads();
    // z/r pre-activations: g[:,0:160] = inp2@Wqa[:,0:160] + h@Uqa[:,0:160] + b
    for (int tile = wave; tile < 20; tile += 8) {
      int mt = tile / 10, nt = tile % 10, ncol = nt * 16 + (lane & 15);
      v8f acc = vzero8();
      for (int kc = 0; kc < 320; kc += 32) {
        v16bf a = ldsA(inp2b, 320, mt * 16 + (lane & 15), kc, lane);
        v16bf b = ldB_global(Wqa, 225, 300, 225, kc, ncol, lane);
        acc = WMMA_BF16(a, b, acc);
      }
      for (int kc = 0; kc < 96; kc += 32) {
        v16bf a = ldsA(hb, 96, mt * 16 + (lane & 15), kc, lane);
        v16bf b = ldB_global(Uqa, 225, 75, 225, kc, ncol, lane);
        acc = WMMA_BF16(a, b, acc);
      }
#pragma unroll
      for (int r = 0; r < 8; ++r) {
        int row = mt * 16 + r + ((lane < 16) ? 0 : 8);
        g[row * 160 + ncol] = acc[r] + bqa[ncol];
      }
    }
    __syncthreads();
    for (int idx = tid; idx < 32 * 75; idx += 256) {
      int b = idx / 75, j = idx % 75;
      float r = sigmf(g[b * 160 + 75 + j]);
      hb[b * 96 + j] = (__bf16)(r * h[b * 80 + j]);
    }
    __syncthreads();
    // candidate pre-act: g[:,80:160] = inp2@Wqa[:,150:225] + (r*h)@Uqa[:,150:225] + b
    for (int tile = wave; tile < 10; tile += 8) {
      int mt = tile / 5, nt = tile % 5, ncol = nt * 16 + (lane & 15);
      v8f acc = vzero8();
      for (int kc = 0; kc < 320; kc += 32) {
        v16bf a = ldsA(inp2b, 320, mt * 16 + (lane & 15), kc, lane);
        v16bf b = ldB_global(Wqa, 225, 300, 225, kc, 150 + ncol, lane);
        acc = WMMA_BF16(a, b, acc);
      }
      for (int kc = 0; kc < 96; kc += 32) {
        v16bf a = ldsA(hb, 96, mt * 16 + (lane & 15), kc, lane);
        v16bf b = ldB_global(Uqa, 225, 75, 225, kc, 150 + ncol, lane);
        acc = WMMA_BF16(a, b, acc);
      }
      if (150 + ncol < 225) {
#pragma unroll
        for (int r = 0; r < 8; ++r) {
          int row = mt * 16 + r + ((lane < 16) ? 0 : 8);
          g[row * 160 + 80 + ncol] = acc[r] + bqa[150 + ncol];
        }
      }
    }
    __syncthreads();
    for (int idx = tid; idx < 32 * 75; idx += 256) {
      int b = idx / 75, j = idx % 75;
      float z  = sigmf(g[b * 160 + j]);
      float hh = tanhf(g[b * 160 + 80 + j]);
      float hn = z * h[b * 80 + j] + (1.f - z) * hh;
      h[b * 80 + j] = hn;
      vP[((size_t)(b * 400 + t)) * 75 + j] = hn;
    }
    __syncthreads();
  }
}

// ---------------------------------------------------------------------------
// Self-attention context (h-independent -> fully parallel): one block per (b,t).
// inp[b,t,:] = [vP[b,t,:], softmax_n(v . tanh(Wpv[b,n]+Wppv[b,t])) @ vP[b,:,:]]
// ---------------------------------------------------------------------------
__global__ __launch_bounds__(256) void sa_context_kernel(
    const float* __restrict__ Wpv, const float* __restrict__ Wppv,
    const float* __restrict__ vP, const float* __restrict__ vvec,
    float* __restrict__ inp) {
  int t = blockIdx.x, b = blockIdx.y, tid = threadIdx.x;
  __shared__ float wt[80], vsh[80], sc[400], red[256];
  if (tid < 75) { wt[tid] = Wppv[((size_t)(b * 400 + t)) * 75 + tid]; vsh[tid] = vvec[tid]; }
  __syncthreads();
  for (int n = tid; n < 400; n += 256) {
    const float* wp = Wpv + ((size_t)(b * 400 + n)) * 75;
    float s = 0.f;
    for (int j = 0; j < 75; ++j) s += vsh[j] * tanhf(wp[j] + wt[j]);
    sc[n] = s;
  }
  __syncthreads();
  float mx = -3.4e38f;
  for (int n = tid; n < 400; n += 256) mx = fmaxf(mx, sc[n]);
  red[tid] = mx; __syncthreads();
  for (int s = 128; s > 0; s >>= 1) { if (tid < s) red[tid] = fmaxf(red[tid], red[tid + s]); __syncthreads(); }
  mx = red[0]; __syncthreads();
  float ps = 0.f;
  for (int n = tid; n < 400; n += 256) { float e = __expf(sc[n] - mx); sc[n] = e; ps += e; }
  red[tid] = ps; __syncthreads();
  for (int s = 128; s > 0; s >>= 1) { if (tid < s) red[tid] += red[tid + s]; __syncthreads(); }
  float inv = 1.f / red[0];
  float* outrow = inp + ((size_t)(b * 400 + t)) * 150;
  for (int d = tid; d < 75; d += 256) {
    float c = 0.f;
    for (int n = 0; n < 400; ++n) c += sc[n] * vP[((size_t)(b * 400 + n)) * 75 + d];
    outrow[75 + d] = c * inv;
    outrow[d] = vP[((size_t)(b * 400 + t)) * 75 + d];
  }
}

// inp = sigmoid(tmp) * inp   (in place)
__global__ void gate_kernel(float* __restrict__ inp, const float* __restrict__ tmp, int n) {
  for (int i = blockIdx.x * blockDim.x + threadIdx.x; i < n; i += gridDim.x * blockDim.x)
    inp[i] = inp[i] * sigmf(tmp[i]);
}

// Question pooling: rQ[b] = softmax_m(v . tanh(WQuQ[b,m] + VQ_r@WQ_v)) @ uQ[b]
__global__ __launch_bounds__(128) void qpool_kernel(
    const float* __restrict__ WQuQ, const float* __restrict__ uQ,
    const float* __restrict__ VQ_r, const float* __restrict__ WQ_v,
    const float* __restrict__ vvec, float* __restrict__ rQ) {
  int b = blockIdx.x, tid = threadIdx.x;
  __shared__ float wqv[80], sc[52];
  for (int j = tid; j < 75; j += 128) {
    float s = 0.f;
    for (int k = 0; k < 75; ++k) s += VQ_r[k] * WQ_v[k * 75 + j];
    wqv[j] = s;
  }
  __syncthreads();
  for (int mq = tid; mq < 50; mq += 128) {
    const float* wq = WQuQ + ((size_t)(b * 50 + mq)) * 75;
    float s = 0.f;
    for (int j = 0; j < 75; ++j) s += vvec[j] * tanhf(wq[j] + wqv[j]);
    sc[mq] = s;
  }
  __syncthreads();
  if (tid == 0) {
    float mx = -3.4e38f;
    for (int q = 0; q < 50; ++q) mx = fmaxf(mx, sc[q]);
    float su = 0.f;
    for (int q = 0; q < 50; ++q) { float e = __expf(sc[q] - mx); sc[q] = e; su += e; }
    float inv = 1.f / su;
    for (int q = 0; q < 50; ++q) sc[q] *= inv;
  }
  __syncthreads();
  for (int d = tid; d < 150; d += 128) {
    float c = 0.f;
    for (int q = 0; q < 50; ++q) c += sc[q] * uQ[((size_t)(b * 50 + q)) * 150 + d];
    rQ[b * 150 + d] = c;
  }
}

// Pointer network: 2 steps, one block per batch element; writes final output.
__global__ __launch_bounds__(256) void pointer_kernel(
    const float* __restrict__ WPh, const float* __restrict__ hP,
    const float* __restrict__ rQ, const float* __restrict__ Wa_h,
    const float* __restrict__ vvec,
    const float* __restrict__ Wp, const float* __restrict__ Up, const float* __restrict__ bp,
    float* __restrict__ out) {
  int b = blockIdx.x, tid = threadIdx.x;
  __shared__ float h[152], ah[80], sc[400], red[256], c[152], mg[452], g1[304], rh[152];
  for (int d = tid; d < 150; d += 256) h[d] = rQ[b * 150 + d];
  __syncthreads();
  for (int step = 0; step < 2; ++step) {
    for (int j = tid; j < 75; j += 256) {
      float s = 0.f;
      for (int k = 0; k < 150; ++k) s += h[k] * Wa_h[k * 75 + j];
      ah[j] = s;
    }
    __syncthreads();
    for (int n = tid; n < 400; n += 256) {
      const float* wp = WPh + ((size_t)(b * 400 + n)) * 75;
      float s = 0.f;
      for (int j = 0; j < 75; ++j) s += vvec[j] * tanhf(wp[j] + ah[j]);
      sc[n] = s;
    }
    __syncthreads();
    float mx = -3.4e38f;
    for (int n = tid; n < 400; n += 256) mx = fmaxf(mx, sc[n]);
    red[tid] = mx; __syncthreads();
    for (int s = 128; s > 0; s >>= 1) { if (tid < s) red[tid] = fmaxf(red[tid], red[tid + s]); __syncthreads(); }
    mx = red[0]; __syncthreads();
    float ps = 0.f;
    for (int n = tid; n < 400; n += 256) { float e = __expf(sc[n] - mx); sc[n] = e; ps += e; }
    red[tid] = ps; __syncthreads();
    for (int s = 128; s > 0; s >>= 1) { if (tid < s) red[tid] += red[tid + s]; __syncthreads(); }
    float inv = 1.f / red[0]; __syncthreads();
    for (int n = tid; n < 400; n += 256) { sc[n] *= inv; out[(size_t)b * 800 + step * 400 + n] = sc[n]; }
    __syncthreads();
    for (int d = tid; d < 150; d += 256) {
      float s = 0.f;
      for (int n = 0; n < 400; ++n) s += sc[n] * hP[((size_t)(b * 400 + n)) * 150 + d];
      c[d] = s;
    }
    __syncthreads();
    for (int jj = tid; jj < 450; jj += 256) {
      float s = bp[jj];
      for (int k = 0; k < 150; ++k) s += c[k] * Wp[k * 450 + jj];
      mg[jj] = s;
    }
    for (int jj = tid; jj < 300; jj += 256) {
      float s = 0.f;
      for (int k = 0; k < 150; ++k) s += h[k] * Up[k * 450 + jj];
      g1[jj] = s;
    }
    __syncthreads();
    for (int j = tid; j < 150; j += 256) { float r = sigmf(mg[150 + j] + g1[150 + j]); rh[j] = r * h[j]; }
    __syncthreads();
    for (int j = tid; j < 150; j += 256) {
      float g2 = 0.f;
      for (int k = 0; k < 150; ++k) g2 += rh[k] * Up[k * 450 + 300 + j];
      float z = sigmf(mg[j] + g1[j]);
      float hh = tanhf(mg[300 + j] + g2);
      h[j] = z * h[j] + (1.f - z) * hh;
    }
    __syncthreads();
  }
}

// ---------------------------------------------------------------------------
// Host orchestration
// ---------------------------------------------------------------------------
extern "C" void kernel_launch(void* const* d_in, const int* in_sizes, int n_in,
                              void* d_out, int out_size, void* d_ws, size_t ws_size,
                              hipStream_t stream) {
  (void)out_size; (void)ws_size; (void)n_in;
  const float* P = (const float*)d_in[0];
  const float* Q = (const float*)d_in[1];
  const float* encP[3][2][3];
  const float* encQ[3][2][3];
  int idx = 2;
  for (int l = 0; l < 3; ++l) for (int d = 0; d < 2; ++d) for (int k = 0; k < 3; ++k)
    encP[l][d][k] = (const float*)d_in[idx++];
  for (int l = 0; l < 3; ++l) for (int d = 0; d < 2; ++d) for (int k = 0; k < 3; ++k)
    encQ[l][d][k] = (const float*)d_in[idx++];

  // Detect flatten order of the nested dicts from leaf sizes.
  int base = 38;
  bool grus_first = (in_sizes[base] == 67500);            // grus leaf (67500) vs shared (75)
  int gb = grus_first ? base : base + 11;
  int sb = grus_first ? base + 12 : base;
  bool s_sorted = (in_sizes[sb + 1] == 5625);             // sorted: WPP_v(5625); insertion: WQ_u(11250)
  bool g_sorted = (in_sizes[gb + 1] == 67500);            // sorted: ptr.U(67500); insertion: qa.U(16875)

  const float *v_, *WQ_u, *WP_u, *WP_v, *W_g1, *W_g2, *WP_h, *Wa_h, *WPP_v, *WQ_v, *VQ_r;
  if (s_sorted) {
    VQ_r = (const float*)d_in[sb + 0]; WPP_v = (const float*)d_in[sb + 1];
    WP_h = (const float*)d_in[sb + 2]; WP_u  = (const float*)d_in[sb + 3];
    WP_v = (const float*)d_in[sb + 4]; WQ_u  = (const float*)d_in[sb + 5];
    WQ_v = (const float*)d_in[sb + 6]; W_g1  = (const float*)d_in[sb + 7];
    W_g2 = (const float*)d_in[sb + 8]; Wa_h  = (const float*)d_in[sb + 9];
    v_   = (const float*)d_in[sb + 10];
  } else {
    v_   = (const float*)d_in[sb + 0]; WQ_u  = (const float*)d_in[sb + 1];
    WP_u = (const float*)d_in[sb + 2]; WP_v  = (const float*)d_in[sb + 3];
    W_g1 = (const float*)d_in[sb + 4]; W_g2  = (const float*)d_in[sb + 5];
    WP_h = (const float*)d_in[sb + 6]; Wa_h  = (const float*)d_in[sb + 7];
    WPP_v= (const float*)d_in[sb + 8]; WQ_v  = (const float*)d_in[sb + 9];
    VQ_r = (const float*)d_in[sb + 10];
  }
  const float *qaW, *qaU, *qab, *sfW, *sfU, *sfb, *sbW, *sbU, *sbb, *ptW, *ptU, *ptb;
  if (g_sorted) { // ptr, qa, sa_b, sa_f
    ptW = (const float*)d_in[gb + 0];  ptU = (const float*)d_in[gb + 1];  ptb = (const float*)d_in[gb + 2];
    qaW = (const float*)d_in[gb + 3];  qaU = (const float*)d_in[gb + 4];  qab = (const float*)d_in[gb + 5];
    sbW = (const float*)d_in[gb + 6];  sbU = (const float*)d_in[gb + 7];  sbb = (const float*)d_in[gb + 8];
    sfW = (const float*)d_in[gb + 9];  sfU = (const float*)d_in[gb + 10]; sfb = (const float*)d_in[gb + 11];
  } else {        // qa, sa_f, sa_b, ptr
    qaW = (const float*)d_in[gb + 0];  qaU = (const float*)d_in[gb + 1];  qab = (const float*)d_in[gb + 2];
    sfW = (const float*)d_in[gb + 3];  sfU = (const float*)d_in[gb + 4];  sfb = (const float*)d_in[gb + 5];
    sbW = (const float*)d_in[gb + 6];  sbU = (const float*)d_in[gb + 7];  sbb = (const float*)d_in[gb + 8];
    ptW = (const float*)d_in[gb + 9];  ptU = (const float*)d_in[gb + 10]; ptb = (const float*)d_in[gb + 11];
  }

  // Workspace layout (floats), with reuse across phases (~78 MB total).
  float* ws = (float*)d_ws;
  float* actP0    = ws;              // 1,920,000
  float* actP1    = ws + 1920000;    // 1,920,000
  float* mf       = ws + 3840000;    // 2,880,000
  float* mb       = ws + 6720000;    // 2,880,000
  float* actQ0    = ws + 9600000;    //   240,000
  float* actQ1    = ws + 9840000;    //   240,000
  float* mqf      = ws + 10080000;   //   360,000
  float* mqb      = ws + 10440000;   //   360,000
  float* WPu_buf  = ws + 10800000;   // 1,200,000 (later reused as WPh)
  float* WQuQ_buf = ws + 12000000;   //   120,000
  float* vP       = ws + 12120000;   // 1,200,000
  float* Wpv_buf  = ws + 13320000;   // 1,200,000 \ reused as hP (1,920,000)
  float* Wppv_buf = ws + 14520000;   // 1,200,000 /
  float* hP       = Wpv_buf;
  float* inp      = ws + 15720000;   // 1,920,000
  float* gatebuf  = ws + 17640000;   // 1,920,000
  float* rQ       = ws + 19560000;   //     4,800

  auto G = [&](const float* A, const float* W, const float* bias, float* C, int R, int K, int N) {
    int tm = (R + 63) / 64, tn = (N + 63) / 64;
    gemm_bias_kernel<<<tm * tn, 256, 0, stream>>>(A, W, bias, C, R, K, N);
  };
  const size_t SCAN_SMEM  = 82944;    // gru_scan_pair dynamic LDS
  const size_t QATTN_SMEM = 294400;   // qattn dynamic LDS

  // --- Question encoder (3 biGRU layers) ---
  const float* xq = Q; int kq = 300;
  float* qo[3] = {actQ0, actQ1, actQ0};
  for (int l = 0; l < 3; ++l) {
    G(xq, encQ[l][0][0], encQ[l][0][2], mqf, 1600, kq, 225);
    G(xq, encQ[l][1][0], encQ[l][1][2], mqb, 1600, kq, 225);
    gru_scan_pair_kernel<<<2, 256, SCAN_SMEM, stream>>>(mqf, encQ[l][0][1], 0, 1, 0,
                                                        mqb, encQ[l][1][1], 1, 1, 75,
                                                        qo[l], 50, 150);
    xq = qo[l]; kq = 150;
  }
  // --- Passage encoder ---
  const float* xp = P; int kp = 300;
  float* po[3] = {actP0, actP1, actP0};
  for (int l = 0; l < 3; ++l) {
    G(xp, encP[l][0][0], encP[l][0][2], mf, 12800, kp, 225);
    G(xp, encP[l][1][0], encP[l][1][2], mb, 12800, kp, 225);
    gru_scan_pair_kernel<<<2, 256, SCAN_SMEM, stream>>>(mf, encP[l][0][1], 0, 1, 0,
                                                        mb, encP[l][1][1], 1, 1, 75,
                                                        po[l], 400, 150);
    xp = po[l]; kp = 150;
  }
  const float* uQ = actQ0;
  const float* uP = actP0;

  // --- Question-attention GRU ---
  G(uQ, WQ_u, nullptr, WQuQ_buf, 1600, 150, 75);
  G(uP, WP_u, nullptr, WPu_buf, 12800, 150, 75);
  qattn_scan_kernel<<<1, 256, QATTN_SMEM, stream>>>(uP, uQ, WPu_buf, WQuQ_buf,
                                                    WP_v, v_, W_g1, qaW, qaU, qab, vP);

  // --- Self-attention: h-independent attention precomputed in parallel ---
  G(vP, WP_v, nullptr, Wpv_buf, 12800, 75, 75);
  G(vP, WPP_v, nullptr, Wppv_buf, 12800, 75, 75);
  sa_context_kernel<<<dim3(400, 32), 256, 0, stream>>>(Wpv_buf, Wppv_buf, vP, v_, inp);
  G(inp, W_g2, nullptr, gatebuf, 12800, 150, 150);
  gate_kernel<<<1920, 256, 0, stream>>>(inp, gatebuf, 12800 * 150);
  G(inp, sfW, sfb, mf, 12800, 150, 225);
  G(inp, sbW, sbb, mb, 12800, 150, 225);
  // sa_f writes cols 0..74 at original time; sa_b reads reversed, writes in scan order
  gru_scan_pair_kernel<<<2, 256, SCAN_SMEM, stream>>>(mf, sfU, 0, 1, 0,
                                                      mb, sbU, 1, 0, 75,
                                                      hP, 400, 150);

  // --- Pooling + pointer network ---
  float* WPh_buf = WPu_buf;
  G(hP, WP_h, nullptr, WPh_buf, 12800, 150, 75);
  qpool_kernel<<<32, 128, 0, stream>>>(WQuQ_buf, uQ, VQ_r, WQ_v, v_, rQ);
  pointer_kernel<<<32, 256, 0, stream>>>(WPh_buf, hP, rQ, Wa_h, v_, ptW, ptU, ptb,
                                         (float*)d_out);
}
